// Discriminative_47433618817341
// MI455X (gfx1250) — matrix operand order repaired and checked by reading, not compiled, not run on hardware
//
#include <hip/hip_runtime.h>
#include <hip/hip_bf16.h>

typedef __attribute__((ext_vector_type(16))) __bf16 v16bf;
typedef __attribute__((ext_vector_type(8)))  float  v8f;

union FragU { v16bf v; uint4 q[2]; };

// ---------------------------------------------------------------------------
// Prep: original -> sq[i] = sum(x^2), bf16 hi/lo split. Also zero the counter.
// ---------------------------------------------------------------------------
__global__ __launch_bounds__(256) void prep_original(
    const float* __restrict__ O, unsigned short* __restrict__ hi,
    unsigned short* __restrict__ lo, float* __restrict__ sq,
    unsigned int* __restrict__ count)
{
    __shared__ float red[256];
    const int row = blockIdx.x, tid = threadIdx.x;
    float ss = 0.0f;
    for (int c = tid; c < 512; c += 256) {
        float x = O[(size_t)row * 512 + c];
        ss += x * x;
        __bf16 h = (__bf16)x;
        __bf16 l = (__bf16)(x - (float)h);
        hi[(size_t)row * 512 + c] = __builtin_bit_cast(unsigned short, h);
        lo[(size_t)row * 512 + c] = __builtin_bit_cast(unsigned short, l);
    }
    red[tid] = ss; __syncthreads();
    for (int s = 128; s > 0; s >>= 1) {
        if (tid < s) red[tid] += red[tid + s];
        __syncthreads();
    }
    if (tid == 0) sq[row] = red[0];
    if (row == 0 && tid == 0) *count = 0u;
}

// ---------------------------------------------------------------------------
// Prep: hidden -> L2-normalized rows, bf16 hi/lo split.
// ---------------------------------------------------------------------------
__global__ __launch_bounds__(256) void prep_hidden(
    const float* __restrict__ H, unsigned short* __restrict__ hi,
    unsigned short* __restrict__ lo)
{
    __shared__ float red[256];
    const int row = blockIdx.x, tid = threadIdx.x;
    float x[4]; float ss = 0.0f;
    #pragma unroll
    for (int q = 0; q < 4; ++q) {
        x[q] = H[(size_t)row * 1024 + tid + q * 256];
        ss += x[q] * x[q];
    }
    red[tid] = ss; __syncthreads();
    for (int s = 128; s > 0; s >>= 1) {
        if (tid < s) red[tid] += red[tid + s];
        __syncthreads();
    }
    const float inv = 1.0f / fmaxf(sqrtf(red[0]), 1e-12f);
    #pragma unroll
    for (int q = 0; q < 4; ++q) {
        float y = x[q] * inv;
        __bf16 h = (__bf16)y;
        __bf16 l = (__bf16)(y - (float)h);
        hi[(size_t)row * 1024 + tid + q * 256] = __builtin_bit_cast(unsigned short, h);
        lo[(size_t)row * 1024 + tid + q * 256] = __builtin_bit_cast(unsigned short, l);
    }
}

// ---------------------------------------------------------------------------
// bf16x3 symmetric GEMM: out = X * X^T (with optional distance epilogue).
// Block tile 64x64, 128 threads = 4 waves, each wave a 32x32 quadrant
// (2x2 grid of 16x16 WMMA tiles). K stepped in 32-element chunks.
// Staging: double-buffered GLOBAL_LOAD_ASYNC_TO_LDS_B128 (ASYNCcnt-tracked),
// overlapped with the WMMA compute of the current chunk.
// MODE 0: store C; MODE 1: store d2 = sq[i] + sq[j] - 2*acc, clamped.
// ---------------------------------------------------------------------------
template <int MODE>
__global__ __launch_bounds__(128) void gemm_bf16x3(
    const unsigned int* __restrict__ hi, const unsigned int* __restrict__ lo,
    int N, int Ddw /* row length in dwords (= D/2) */,
    const float* __restrict__ sq, float* __restrict__ out)
{
    // rows padded to 20 dwords (16B-aligned rows, bank-conflict-free gathers)
    __shared__ unsigned int sLds[2][4][64 * 20];

    const int tid  = threadIdx.x;
    const int bn   = blockIdx.x, bm = blockIdx.y;
    const int lane = tid & 31,  wv = tid >> 5;
    const int wqm  = wv >> 1,   wqn = wv & 1;
    const int hg   = lane >> 4, lm  = lane & 15;

    v8f acc[2][2] = {};

    const size_t aBase = (size_t)(bm * 64) * Ddw;
    const size_t bBase = (size_t)(bn * 64) * Ddw;

    // Issue async global->LDS copies of one 4x(64x32bf16) chunk into buffer b.
    auto stage = [&](int k, int b) {
        const unsigned int* srcs[4] = { hi + aBase + k, lo + aBase + k,
                                        hi + bBase + k, lo + bBase + k };
        #pragma unroll
        for (int mat = 0; mat < 4; ++mat) {
            const unsigned int* src = srcs[mat];
            #pragma unroll
            for (int t = 0; t < 2; ++t) {
                const int q  = tid + t * 128;     // 0..255: b128 chunk id in mat
                const int r  = q >> 2, c4 = q & 3;
                const unsigned ldsOff =
                    (unsigned)(uintptr_t)(&sLds[b][mat][r * 20 + c4 * 4]);
                const unsigned long long gp =
                    (unsigned long long)(uintptr_t)(src + (size_t)r * Ddw + c4 * 4);
                asm volatile("global_load_async_to_lds_b128 %0, %1, off"
                             :: "v"(ldsOff), "v"(gp) : "memory");
            }
        }
    };

    const int nIters = Ddw >> 4;                  // 16 dwords = 32 bf16 per step
    stage(0, 0);
    asm volatile("s_wait_asynccnt 0x0" ::: "memory");
    __syncthreads();

    for (int it = 0; it < nIters; ++it) {
        const int cur = it & 1;
        if (it + 1 < nIters) stage((it + 1) << 4, cur ^ 1); // prefetch next chunk

        // Gather WMMA fragments (CDNA5 16-bit A 16x32 / B 32x16 layouts).
        FragU ah[2], al[2], bh[2], bl[2];
        #pragma unroll
        for (int mt = 0; mt < 2; ++mt) {
            const int rloc = wqm * 32 + mt * 16 + lm;
            const unsigned int* bH = &sLds[cur][0][rloc * 20];
            const unsigned int* bL = &sLds[cur][1][rloc * 20];
            // A: lanes<16 -> Kpairs {0..3, 8..11}; lanes>=16 -> {4..7, 12..15}
            ah[mt].q[0] = *(const uint4*)(bH + hg * 4);
            ah[mt].q[1] = *(const uint4*)(bH + 8 + hg * 4);
            al[mt].q[0] = *(const uint4*)(bL + hg * 4);
            al[mt].q[1] = *(const uint4*)(bL + 8 + hg * 4);
        }
        #pragma unroll
        for (int nt = 0; nt < 2; ++nt) {
            const int rloc = wqn * 32 + nt * 16 + lm;
            const unsigned int* bH = &sLds[cur][2][rloc * 20];
            const unsigned int* bL = &sLds[cur][3][rloc * 20];
            // B: lanes<16 -> Kpairs {0..3, 8..11}; lanes>=16 -> {4..7, 12..15}
            bh[nt].q[0] = *(const uint4*)(bH + hg * 4);
            bh[nt].q[1] = *(const uint4*)(bH + 8 + hg * 4);
            bl[nt].q[0] = *(const uint4*)(bL + hg * 4);
            bl[nt].q[1] = *(const uint4*)(bL + 8 + hg * 4);
        }

        #pragma unroll
        for (int mt = 0; mt < 2; ++mt)
            #pragma unroll
            for (int nt = 0; nt < 2; ++nt) {
                acc[mt][nt] = __builtin_amdgcn_wmma_f32_16x16x32_bf16(
                    false, ah[mt].v, false, bh[nt].v, (short)0, acc[mt][nt], false, false);
                acc[mt][nt] = __builtin_amdgcn_wmma_f32_16x16x32_bf16(
                    false, al[mt].v, false, bh[nt].v, (short)0, acc[mt][nt], false, false);
                acc[mt][nt] = __builtin_amdgcn_wmma_f32_16x16x32_bf16(
                    false, ah[mt].v, false, bl[nt].v, (short)0, acc[mt][nt], false, false);
            }

        asm volatile("s_wait_asynccnt 0x0" ::: "memory");
        __syncthreads();
    }

    // Epilogue. C/D layout: VGPR r -> row (r + 8*hg), col = lane%16.
    #pragma unroll
    for (int mt = 0; mt < 2; ++mt) {
        #pragma unroll
        for (int nt = 0; nt < 2; ++nt) {
            const int gcol = bn * 64 + wqn * 32 + nt * 16 + lm;
            const float scol = (MODE == 1) ? sq[gcol] : 0.0f;
            #pragma unroll
            for (int r = 0; r < 8; ++r) {
                const int grow = bm * 64 + wqm * 32 + mt * 16 + hg * 8 + r;
                float v = acc[mt][nt][r];
                if (MODE == 1) {
                    v = sq[grow] + scol - 2.0f * v;
                    v = fmaxf(v, 0.0f);
                }
                out[(size_t)grow * N + gcol] = v;
            }
        }
    }
}

// ---------------------------------------------------------------------------
// Per-row 10th-smallest d^2 (kNN threshold, K=10 incl. self).
// ---------------------------------------------------------------------------
__device__ __forceinline__ void insert10(float (&v)[10], float x)
{
    if (x < v[9]) {
        v[9] = x;
        #pragma unroll
        for (int p = 9; p > 0; --p) { // one bubble pass restores sortedness
            float a = v[p - 1], b = v[p];
            v[p - 1] = fminf(a, b);
            v[p]     = fmaxf(a, b);
        }
    }
}

__global__ __launch_bounds__(256) void topk_thresh(
    const float* __restrict__ d2, int N, float* __restrict__ T2)
{
    __shared__ float l1[256 * 10];
    __shared__ float l2[32 * 10];
    const int row = blockIdx.x, tid = threadIdx.x;
    const float* rp = d2 + (size_t)row * N;

    float v[10];
    #pragma unroll
    for (int p = 0; p < 10; ++p) v[p] = 3.4e38f;
    for (int j = tid; j < N; j += 256) insert10(v, rp[j]);
    #pragma unroll
    for (int p = 0; p < 10; ++p) l1[tid * 10 + p] = v[p];
    __syncthreads();

    if (tid < 32) {
        #pragma unroll
        for (int p = 0; p < 10; ++p) v[p] = 3.4e38f;
        for (int s = 0; s < 8; ++s)
            for (int p = 0; p < 10; ++p) insert10(v, l1[(tid * 8 + s) * 10 + p]);
        #pragma unroll
        for (int p = 0; p < 10; ++p) l2[tid * 10 + p] = v[p];
    }
    __syncthreads();

    if (tid == 0) {
        #pragma unroll
        for (int p = 0; p < 10; ++p) v[p] = 3.4e38f;
        for (int s = 0; s < 32; ++s)
            for (int p = 0; p < 10; ++p) insert10(v, l2[s * 10 + p]);
        T2[row] = v[9]; // 10th smallest (rank K-1)
    }
}

// ---------------------------------------------------------------------------
// a_sum = #{(i,j) : d2[i][j] <= max(T2[i], T2[j])}  (symmetrized kNN mask)
// ---------------------------------------------------------------------------
__global__ __launch_bounds__(256) void count_anchor(
    const float* __restrict__ d2, const float* __restrict__ T2, int N,
    unsigned int* __restrict__ count)
{
    __shared__ unsigned int red[256];
    const int i = blockIdx.x, tid = threadIdx.x;
    const float ti = T2[i];
    const float* rp = d2 + (size_t)i * N;
    unsigned int c = 0;
    for (int j = tid; j < N; j += 256)
        c += (rp[j] <= fmaxf(ti, T2[j])) ? 1u : 0u;
    red[tid] = c; __syncthreads();
    for (int s = 128; s > 0; s >>= 1) {
        if (tid < s) red[tid] += red[tid + s];
        __syncthreads();
    }
    if (tid == 0) atomicAdd(count, red[0]);
}

// ---------------------------------------------------------------------------
// In-place rewrite of both d_out halves with the scaled outputs.
// ---------------------------------------------------------------------------
__global__ __launch_bounds__(256) void finalize(
    const float* __restrict__ T2, const unsigned int* __restrict__ count,
    int N, float* __restrict__ out)
{
    const int i = blockIdx.x, tid = threadIdx.x;
    const float a_sum = (float)(*count);
    const float Nw = 0.5f / a_sum;                        // (1-ALPHA)/a_sum
    const float nn = (float)N * (float)N;
    const float Nb = 1.0f / (nn - 2.0f * a_sum);          // 1/(sum(non)-a_sum)
    const float ti = T2[i];
    float* Crow = out + (size_t)i * N;                    // cosine sim (scratch)
    float* Drow = out + (size_t)N * N + (size_t)i * N;    // d^2 (scratch)
    for (int j = tid; j < N; j += 256) {
        const float dv = Drow[j];
        const float cv = Crow[j];
        const float anc = (dv <= fmaxf(ti, T2[j])) ? 1.0f : 0.0f;
        Crow[j] = Nb * cv * (1.0f - anc); // non_anchor_out
        Drow[j] = Nw * cv * anc;          // anchor_out
    }
}

// ---------------------------------------------------------------------------
extern "C" void kernel_launch(void* const* d_in, const int* in_sizes, int n_in,
                              void* d_out, int out_size, void* d_ws, size_t ws_size,
                              hipStream_t stream)
{
    (void)n_in; (void)out_size; (void)ws_size;
    const float* O = (const float*)d_in[0];   // [N,512]
    const float* H = (const float*)d_in[1];   // [N,1024]
    float* out = (float*)d_out;               // [2,N,N]
    const int N = in_sizes[0] / 512;          // 4096

    // Workspace layout
    char* ws = (char*)d_ws;
    float* sq = (float*)ws;
    float* T2 = (float*)(ws + (size_t)N * 4);
    unsigned int* count = (unsigned int*)(ws + 2 * (size_t)N * 4);
    size_t off = 2 * (size_t)N * 4 + 256;
    unsigned short* Ohi = (unsigned short*)(ws + off); off += (size_t)N * 512 * 2;
    unsigned short* Olo = (unsigned short*)(ws + off); off += (size_t)N * 512 * 2;
    unsigned short* Hhi = (unsigned short*)(ws + off); off += (size_t)N * 1024 * 2;
    unsigned short* Hlo = (unsigned short*)(ws + off);

    float* Cmat = out;                        // cosine sim -> becomes non_anchor_out
    float* Dmat = out + (size_t)N * N;        // d^2 scratch -> becomes anchor_out

    prep_original<<<N, 256, 0, stream>>>(O, Ohi, Olo, sq, count);
    prep_hidden  <<<N, 256, 0, stream>>>(H, Hhi, Hlo);

    dim3 g(N / 64, N / 64);
    gemm_bf16x3<0><<<g, 128, 0, stream>>>((const unsigned int*)Hhi, (const unsigned int*)Hlo,
                                          N, 1024 / 2, nullptr, Cmat);
    gemm_bf16x3<1><<<g, 128, 0, stream>>>((const unsigned int*)Ohi, (const unsigned int*)Olo,
                                          N, 512 / 2, sq, Dmat);

    topk_thresh <<<N, 256, 0, stream>>>(Dmat, N, T2);
    count_anchor<<<N, 256, 0, stream>>>(Dmat, T2, N, count);
    finalize    <<<N, 256, 0, stream>>>(T2, count, N, out);
}
